// AdditiveAttention_63161789055237
// MI455X (gfx1250) — matrix operand order repaired and checked by reading, not compiled, not run on hardware
//
#include <hip/hip_runtime.h>

// Problem constants (reference: BS, J, T, D = 2, 512, 256, 512)
constexpr int kBS = 2;
constexpr int kJ  = 512;
constexpr int kT  = 256;
constexpr int kD  = 512;

typedef __attribute__((ext_vector_type(16))) __bf16 v16bf;
typedef __attribute__((ext_vector_type(8)))  float  v8f;

// ---------------------------------------------------------------------------
// gfx1250 native tanh (v_tanh_f32, TRANS op) with exp2-based fallback.
// ---------------------------------------------------------------------------
__device__ __forceinline__ float fast_tanh(float x) {
#if __has_builtin(__builtin_amdgcn_tanhf)
  return __builtin_amdgcn_tanhf(x);
#else
  float ax = __builtin_fabsf(x);
  float t  = __builtin_amdgcn_exp2f(ax * -2.885390081777927f);   // e^{-2|x|}
  float r  = (1.0f - t) * __builtin_amdgcn_rcpf(1.0f + t);
  return __builtin_copysignf(r, x);
#endif
}

// ---------------------------------------------------------------------------
// WMMA bf16 operand loaders (wave32, contiguous k order, same for A and B).
//
// A: 16(M) x 32(K), source row-major with leading dim ldm.
//    lanes 0-15 hold row M=lane,   K in {0..15}
//    lanes 16-31 hold row M=lane-16, K in {16..31}
// ---------------------------------------------------------------------------
__device__ __forceinline__ v16bf load_a_bf16(const float* __restrict__ base,
                                             int ldm, int lane) {
  const int row = lane & 15, half = lane >> 4;
  const float* p = base + (size_t)row * ldm + half * 16;
  v16bf a;
#pragma unroll
  for (int i = 0; i < 16; ++i) {
    a[i]     = (__bf16)p[i];
  }
  return a;
}

// B: 32(K) x 16(N) where B[k][n] = base[n*ldm + k]  (transpose of row-major;
// per-lane reads are contiguous in k).  lanes 0-15: col n=lane, K=0..15;
// lanes 16-31: col n=lane-16, K=16..31.
__device__ __forceinline__ v16bf load_bT_bf16(const float* __restrict__ base,
                                              int ldm, int lane) {
  const int col = lane & 15, half = lane >> 4;
  const float* p = base + (size_t)col * ldm + half * 16;
  v16bf b;
#pragma unroll
  for (int i = 0; i < 16; ++i) b[i] = (__bf16)p[i];
  return b;
}

// B: 32(K) x 16(N) where B[k][n] = base[k*ldm + n]  (row-major source).
__device__ __forceinline__ v16bf load_b_bf16(const float* __restrict__ base,
                                             int ldm, int lane) {
  const int col = lane & 15, half = lane >> 4;
  const float* p = base + (size_t)(half * 16) * ldm + col;
  v16bf b;
#pragma unroll
  for (int i = 0; i < 16; ++i) b[i] = (__bf16)p[(size_t)i * ldm];
  return b;
}

// C/D 16x16 f32: VGPR r -> (M = half*8 + r, N = lane&15)
__device__ __forceinline__ void store_c_f32(float* __restrict__ base, int ldm,
                                            int lane, v8f c) {
  const int col = lane & 15, half = lane >> 4;
#pragma unroll
  for (int r = 0; r < 8; ++r)
    base[(size_t)(half * 8 + r) * ldm + col] = c[r];
}

// ---------------------------------------------------------------------------
// Stage 1: kkT[b][e][j] = sum_d Wk[e][d] * k[b][j][d]   (stored TRANSPOSED so
// the energy kernel gets lane-contiguous j reads).
// One wave per 16(e)x16(j) tile; K-loop = 512/32 = 16 WMMAs.
// ---------------------------------------------------------------------------
__global__ __launch_bounds__(256) void proj_k_kernel(
    const float* __restrict__ kin, const float* __restrict__ Wk,
    float* __restrict__ kkT) {
  const int lane = threadIdx.x & 31;
  const int wave = blockIdx.x * 8 + (threadIdx.x >> 5);
  constexpr int tiles_per_b = (kD / 16) * (kJ / 16);      // 1024
  const int b  = wave / tiles_per_b;
  const int r  = wave % tiles_per_b;
  const int e0 = (r / (kJ / 16)) * 16;
  const int j0 = (r % (kJ / 16)) * 16;
  const float* kb = kin + (size_t)b * kJ * kD;

  v8f acc = {};
  for (int d0 = 0; d0 < kD; d0 += 32) {
    v16bf a  = load_a_bf16(Wk + (size_t)e0 * kD + d0, kD, lane);
    v16bf bm = load_bT_bf16(kb + (size_t)j0 * kD + d0, kD, lane);
    acc = __builtin_amdgcn_wmma_f32_16x16x32_bf16(
        false, a, false, bm, (short)0, acc, false, false);
  }
  store_c_f32(kkT + ((size_t)b * kD + e0) * kJ + j0, kJ, lane, acc);
}

// ---------------------------------------------------------------------------
// Stage 2: qq[b][t][e] = sum_d q[b][t][d] * Wq[e][d] + bq[e]
// ---------------------------------------------------------------------------
__global__ __launch_bounds__(256) void proj_q_kernel(
    const float* __restrict__ qin, const float* __restrict__ Wq,
    const float* __restrict__ bq, float* __restrict__ qq) {
  const int lane = threadIdx.x & 31;
  const int wave = blockIdx.x * 8 + (threadIdx.x >> 5);
  constexpr int tiles_per_b = (kT / 16) * (kD / 16);      // 512
  const int b  = wave / tiles_per_b;
  const int r  = wave % tiles_per_b;
  const int t0 = (r / (kD / 16)) * 16;
  const int e0 = (r % (kD / 16)) * 16;
  const float* qb = qin + (size_t)b * kT * kD;

  v8f acc = {};
  for (int d0 = 0; d0 < kD; d0 += 32) {
    v16bf a  = load_a_bf16(qb + (size_t)t0 * kD + d0, kD, lane);
    v16bf bm = load_bT_bf16(Wq + (size_t)e0 * kD + d0, kD, lane);
    acc = __builtin_amdgcn_wmma_f32_16x16x32_bf16(
        false, a, false, bm, (short)0, acc, false, false);
  }
  const int col = lane & 15, half = lane >> 4;
  const float bias = bq[e0 + col];
  float* out = qq + ((size_t)b * kT + t0) * kD + e0;
#pragma unroll
  for (int rr = 0; rr < 8; ++rr)
    out[(size_t)(half * 8 + rr) * kD + col] = acc[rr] + bias;
}

// ---------------------------------------------------------------------------
// Stage 3: Bahdanau energy (the hot loop: 134M tanh).
// One block per (b,t); 512 threads = all j.  qq row + we broadcast from LDS;
// kkT read lane-contiguous (stride J per e step).  Masked score written
// straight into the alphas half of d_out, layout [b][t][j].
// ---------------------------------------------------------------------------
__global__ __launch_bounds__(512) void energy_kernel(
    const float* __restrict__ kkT, const float* __restrict__ qq,
    const float* __restrict__ we, const int* __restrict__ mask,
    float* __restrict__ scores) {
  __shared__ __align__(16) float s_q[kD];
  __shared__ __align__(16) float s_w[kD];
  const int bt = blockIdx.x;               // 0 .. BS*T-1
  const int b  = bt / kT;
  const int t  = bt % kT;
  const int j  = threadIdx.x;              // 0 .. 511

  s_q[j] = qq[((size_t)b * kT + t) * kD + j];
  s_w[j] = we[j];
  __syncthreads();

  const float* kc = kkT + (size_t)b * kD * kJ + j;   // kkT[b][e][j], stride kJ
  float acc = 0.0f;
#pragma unroll 8
  for (int e = 0; e < kD; ++e)
    acc += s_w[e] * fast_tanh(kc[(size_t)e * kJ] + s_q[e]);

  if (mask[b * kJ + j] == 0) acc = -1.0e9f;
  scores[((size_t)b * kT + t) * kJ + j] = acc;
}

// ---------------------------------------------------------------------------
// Stage 4: in-place softmax over J (512) for each of the BS*T = 512 rows.
// One wave32 per row; float4 vector accesses; shfl_xor reductions.
// ---------------------------------------------------------------------------
__global__ __launch_bounds__(256) void softmax_kernel(float* __restrict__ sc) {
  const int lane = threadIdx.x & 31;
  const int row  = blockIdx.x * 8 + (threadIdx.x >> 5);
  float* p = sc + (size_t)row * kJ;

  float4 x[4];
  float mx = -3.4e38f;
#pragma unroll
  for (int c = 0; c < 4; ++c) {
    x[c] = reinterpret_cast<const float4*>(p)[c * 32 + lane];
    mx = fmaxf(mx, fmaxf(fmaxf(x[c].x, x[c].y), fmaxf(x[c].z, x[c].w)));
  }
#pragma unroll
  for (int s = 16; s > 0; s >>= 1) mx = fmaxf(mx, __shfl_xor(mx, s, 32));

  constexpr float kLog2e = 1.4426950408889634f;
  float sum = 0.0f;
#pragma unroll
  for (int c = 0; c < 4; ++c) {
    x[c].x = __builtin_amdgcn_exp2f((x[c].x - mx) * kLog2e);
    x[c].y = __builtin_amdgcn_exp2f((x[c].y - mx) * kLog2e);
    x[c].z = __builtin_amdgcn_exp2f((x[c].z - mx) * kLog2e);
    x[c].w = __builtin_amdgcn_exp2f((x[c].w - mx) * kLog2e);
    sum += x[c].x + x[c].y + x[c].z + x[c].w;
  }
#pragma unroll
  for (int s = 16; s > 0; s >>= 1) sum += __shfl_xor(sum, s, 32);

  const float inv = __builtin_amdgcn_rcpf(sum);
#pragma unroll
  for (int c = 0; c < 4; ++c) {
    x[c].x *= inv; x[c].y *= inv; x[c].z *= inv; x[c].w *= inv;
    reinterpret_cast<float4*>(p)[c * 32 + lane] = x[c];
  }
}

// ---------------------------------------------------------------------------
// Stage 5: context[b][t][d] = sum_j alphas[b][t][j] * v[b][j][d]  (bf16 WMMA)
// ---------------------------------------------------------------------------
__global__ __launch_bounds__(256) void context_kernel(
    const float* __restrict__ alphas, const float* __restrict__ vin,
    float* __restrict__ ctx) {
  const int lane = threadIdx.x & 31;
  const int wave = blockIdx.x * 8 + (threadIdx.x >> 5);
  constexpr int tiles_per_b = (kT / 16) * (kD / 16);      // 512
  const int b  = wave / tiles_per_b;
  const int r  = wave % tiles_per_b;
  const int t0 = (r / (kD / 16)) * 16;
  const int d0 = (r % (kD / 16)) * 16;
  const float* ab = alphas + (size_t)b * kT * kJ;
  const float* vb = vin + (size_t)b * kJ * kD;

  v8f acc = {};
  for (int j0 = 0; j0 < kJ; j0 += 32) {
    v16bf a  = load_a_bf16(ab + (size_t)t0 * kJ + j0, kJ, lane);
    v16bf bm = load_b_bf16(vb + (size_t)j0 * kD + d0, kD, lane);
    acc = __builtin_amdgcn_wmma_f32_16x16x32_bf16(
        false, a, false, bm, (short)0, acc, false, false);
  }
  store_c_f32(ctx + ((size_t)b * kT + t0) * kD + d0, kD, lane, acc);
}

// ---------------------------------------------------------------------------
// Host launcher
// ---------------------------------------------------------------------------
extern "C" void kernel_launch(void* const* d_in, const int* in_sizes, int n_in,
                              void* d_out, int out_size, void* d_ws,
                              size_t ws_size, hipStream_t stream) {
  (void)in_sizes; (void)n_in; (void)out_size; (void)ws_size;

  const float* kin  = (const float*)d_in[0];   // [BS, J, D]
  const float* vin  = (const float*)d_in[1];   // [BS, J, D]
  const float* qin  = (const float*)d_in[2];   // [BS, T, D]
  const int*   mask = (const int*)d_in[3];     // [BS, J, 1]
  const float* Wq   = (const float*)d_in[4];   // [D, D]
  const float* bq   = (const float*)d_in[5];   // [D]
  const float* Wk   = (const float*)d_in[6];   // [D, D]
  const float* we   = (const float*)d_in[7];   // [D]

  float* ctx    = (float*)d_out;                         // [BS, T, D]
  float* alphas = ctx + (size_t)kBS * kT * kD;           // [BS, T, J]

  float* kkT = (float*)d_ws;                             // [BS, D, J]  2 MB
  float* qq  = kkT + (size_t)kBS * kD * kJ;              // [BS, T, D]  1 MB

  // 2048 tiles / 8 waves per block
  proj_k_kernel<<<256, 256, 0, stream>>>(kin, Wk, kkT);
  // 1024 tiles
  proj_q_kernel<<<128, 256, 0, stream>>>(qin, Wq, bq, qq);
  // one block per (b,t)
  energy_kernel<<<kBS * kT, 512, 0, stream>>>(kkT, qq, we, mask, alphas);
  // 512 rows, 8 waves per block
  softmax_kernel<<<(kBS * kT) / 8, 256, 0, stream>>>(alphas);
  // 1024 tiles
  context_kernel<<<128, 256, 0, stream>>>(alphas, vin, ctx);
}
